// FastSpeech_24300924961214
// MI455X (gfx1250) — matrix operand order repaired and checked by reading, not compiled, not run on hardware
//
#include <hip/hip_runtime.h>
#include <hip/hip_bf16.h>

// ---------------------------------------------------------------------------
// FastSpeech forward for MI455X (gfx1250, wave32, WMMA).
// All GEMM-shaped work (QKV/O projections, conv1 taps, conv2, mel head) runs
// through v_wmma_f32_16x16x32_bf16 with fp32 accumulation.
//
// Shape guarantees exploited:
//   M % 64 == 0  (2048 enc rows, 16384 dec rows)
//   N % 16 == 0 and N % 4 == 0  (256, 1024, 80)
//   K % 32 == 0  (256, 1024)
// ---------------------------------------------------------------------------

#define D_    256
#define H_    2
#define DK_   128
#define FF_   1024
#define NL_   4
#define B_    16
#define LSRC_ 128
#define MMAX_ 1024
#define K1_   9
#define P1_   4
#define NMELS_ 80

#define ROWS_ENC (B_ * LSRC_)   // 2048
#define ROWS_DEC (B_ * MMAX_)   // 16384

typedef __attribute__((ext_vector_type(16))) __bf16 v16bf;
typedef __attribute__((ext_vector_type(8)))  __bf16 v8bf;
typedef __attribute__((ext_vector_type(2)))  __bf16 v2bf;
typedef __attribute__((ext_vector_type(8)))  float  v8f;

union F16x16 { v16bf v; v8bf h[2]; };

__device__ __forceinline__ v8bf cvt8(float4 a, float4 b)
{
    v8bf r;
    r[0] = (__bf16)a.x; r[1] = (__bf16)a.y; r[2] = (__bf16)a.z; r[3] = (__bf16)a.w;
    r[4] = (__bf16)b.x; r[5] = (__bf16)b.y; r[6] = (__bf16)b.z; r[7] = (__bf16)b.w;
    return r;
}

// ---------------------------------------------------------------------------
// WMMA GEMM:  C[M,N] = act( sum_t A_shift(t)[M,K] * B_t[K,N] + bias )
//   * 128 threads = 4 wave32s. Block tile = 64 rows x 64 cols.
//     Wave w owns M-tile (bx*64 + 16w); all waves share the 64-wide N panel
//     with 4 accumulators per wave -> 4 WMMA per k-chunk, A fragment reused 4x.
//   * A fragments loaded straight from global as 4x b128 per lane per chunk
//     (ISA 16-bit A layout = two contiguous 8-float row segments per lane).
//   * B panel staged in LDS transposed+bf16 ([n][k]); each thread owns a
//     k-pair x 8 n's so staging stores are packed ds_store_b32.
//     Fragment = 2x ds b128 per sub-tile.
//   * B element (k,n) at Bmat[t*tapBStep + k*bs_k + n*bs_n].
//   * Conv taps: A row shift = tapShift0 + t, zero-padded per Lrows batch;
//     handled branchlessly (clamped pointer + fragment zeroing).
//   * N boundary (mel head, N=80): staging columns clamped (N%4==0), epilogue
//     guarded per 16-wide sub-tile (uniform branch). No divergent memory ops.
// ---------------------------------------------------------------------------
template <bool RELU, bool BVEC>
__global__ __launch_bounds__(128)
void gemm_wmma_kernel(const float* __restrict__ A, int lda, int Lrows,
                      const float* __restrict__ Bmat, long bs_k, long bs_n,
                      const float* __restrict__ bias,
                      const float* __restrict__ resid,
                      const float* __restrict__ rowmask,
                      float* __restrict__ C, int ldc,
                      int N, int K, int ntaps, int tapShift0, long tapBStep)
{
    __shared__ __align__(16) __bf16 Bs[64][40];   // [n][k], 80B row stride (conflict-free)

    const int tid  = threadIdx.x;
    const int wave = tid >> 5;
    const int lane = tid & 31;
    const int tileM = blockIdx.x * 64 + wave * 16;
    const int tileN = blockIdx.y * 64;
    const int rem   = N - tileN;                  // valid columns in panel (>=16)

    // ---- B staging mapping: thread owns k-pair (2*kp, 2*kp+1) x 8 n's ----
    const int kp  = tid & 15;                     // k-pair index 0..15
    const int bn0 = (tid >> 4) * 8;               // 0,8,...,56
    // clamped float4 group bases (N % 4 == 0 -> groups are all-or-nothing)
    const int g0 = (bn0 + 4     <= rem) ? bn0     : (rem - 4);
    const int g1 = (bn0 + 8     <= rem) ? bn0 + 4 : (rem - 4);

    // ---- A fragment mapping ----------------------------------------------
    const int half = lane >> 4;
    const int am   = lane & 15;
    const int arow = tileM + am;
    const int nr   = lane & 15;

    v8f acc[4] = {};

    for (int t = 0; t < ntaps; ++t) {
        const int shift = tapShift0 + t;
        bool rowok = true;
        if (Lrows > 0) {
            const int pos = arow % Lrows;
            rowok = (pos + shift >= 0) && (pos + shift < Lrows);
        }
        // clamped pointer: always reads a valid row; junk discarded below
        const float* aprow = A + (long)(rowok ? (arow + shift) : arow) * lda;
        const float* bbase = Bmat + (long)t * tapBStep + (long)tileN * bs_n;

        for (int k0 = 0; k0 < K; k0 += 32) {
            // ---- stage B panel (block cooperative, transposed bf16) -------
            {
                const float* bp0 = bbase + (long)(k0 + 2 * kp) * bs_k;       // k row 0
                const float* bp1 = bp0 + bs_k;                               // k row 1
                float r0[8], r1[8];
                if (BVEC) {                       // bs_n == 1
                    const float4 f00 = *(const float4*)(bp0 + g0);
                    const float4 f01 = *(const float4*)(bp0 + g1);
                    const float4 f10 = *(const float4*)(bp1 + g0);
                    const float4 f11 = *(const float4*)(bp1 + g1);
                    r0[0]=f00.x; r0[1]=f00.y; r0[2]=f00.z; r0[3]=f00.w;
                    r0[4]=f01.x; r0[5]=f01.y; r0[6]=f01.z; r0[7]=f01.w;
                    r1[0]=f10.x; r1[1]=f10.y; r1[2]=f10.z; r1[3]=f10.w;
                    r1[4]=f11.x; r1[5]=f11.y; r1[6]=f11.z; r1[7]=f11.w;
                } else {
                    #pragma unroll
                    for (int i = 0; i < 8; ++i) {
                        const int n  = bn0 + i;
                        const int nc = (n < rem) ? n : (rem - 1);
                        r0[i] = bp0[(long)nc * bs_n];
                        r1[i] = bp1[(long)nc * bs_n];
                    }
                }
                #pragma unroll
                for (int i = 0; i < 8; ++i) {     // packed bf16 pair -> ds_store_b32
                    v2bf p;
                    p[0] = (__bf16)r0[i];
                    p[1] = (__bf16)r1[i];
                    *(v2bf*)&Bs[bn0 + i][2 * kp] = p;
                }
            }
            // ---- A fragment direct from global (4x b128) ------------------
            if (k0 + 32 < K)
                __builtin_prefetch(aprow + k0 + 32, 0, 1);   // uniform branch
            const float4* ap = (const float4*)(aprow + k0);
            const float4 a0 = ap[2 * half];
            const float4 a1 = ap[2 * half + 1];
            const float4 a2 = ap[4 + 2 * half];
            const float4 a3 = ap[4 + 2 * half + 1];
            F16x16 af;
            af.h[0] = cvt8(a0, a1);
            af.h[1] = cvt8(a2, a3);
            if (!rowok) {                 // zero contribution (cndmask, no branchy mem)
                const v8bf z = {};
                af.h[0] = z;
                af.h[1] = z;
            }
            __syncthreads();
            // ---- 4 sub-tiles: B fragment = 2x contiguous 16B LDS reads ----
            #pragma unroll
            for (int j = 0; j < 4; ++j) {
                F16x16 bfr;
                bfr.h[0] = *(const v8bf*)&Bs[j * 16 + nr][8 * half];
                bfr.h[1] = *(const v8bf*)&Bs[j * 16 + nr][16 + 8 * half];
                acc[j] = __builtin_amdgcn_wmma_f32_16x16x32_bf16(
                             false, af.v, false, bfr.v, (short)0, acc[j], false, false);
            }
            __syncthreads();
        }
    }

    // ---- epilogue: lane layout — VGPR r -> row (r + 8*half), col (lane&15)
    #pragma unroll
    for (int j = 0; j < 4; ++j) {
        if (tileN + j * 16 < N) {                 // uniform guard (N % 16 == 0)
            const int gn = tileN + j * 16 + nr;
            const float bv = bias ? bias[gn] : 0.0f;
            #pragma unroll
            for (int r = 0; r < 8; ++r) {
                const int gm = tileM + r + 8 * half;
                float v = acc[j][r] + bv;
                if (resid)   v += resid[(long)gm * ldc + gn];
                if (RELU)    v = v > 0.0f ? v : 0.0f;
                if (rowmask) v *= rowmask[gm];
                C[(long)gm * ldc + gn] = v;
            }
        }
    }
}

// ---------------------------------------------------------------------------
// Masked softmax attention: one block per (b, h, query). DK=128.
// ---------------------------------------------------------------------------
__global__ __launch_bounds__(128)
void attention_kernel(const float* __restrict__ q, const float* __restrict__ k,
                      const float* __restrict__ v, const float* __restrict__ rowmask,
                      float* __restrict__ o, int L)
{
    __shared__ float qs[DK_];
    __shared__ float sc[MMAX_];
    __shared__ float red[128];

    int idx = blockIdx.x;
    const int qi = idx % L; idx /= L;
    const int h  = idx % H_; idx /= H_;
    const int b  = idx;
    const int tid = threadIdx.x;

    const long rowQ = (long)(b * L + qi) * (H_ * DK_) + h * DK_;
    for (int d = tid; d < DK_; d += 128) qs[d] = q[rowQ + d];
    __syncthreads();

    const float scale = rsqrtf((float)DK_);
    for (int kk = tid; kk < L; kk += 128) {
        const float* krow = k + (long)(b * L + kk) * (H_ * DK_) + h * DK_;
        float s = 0.0f;
        for (int d = 0; d < DK_; ++d) s += qs[d] * krow[d];
        const bool valid = rowmask[b * L + kk] > 0.5f;
        sc[kk] = valid ? s * scale : -__builtin_inff();
    }
    __syncthreads();

    float m = -__builtin_inff();
    for (int kk = tid; kk < L; kk += 128) m = fmaxf(m, sc[kk]);
    red[tid] = m; __syncthreads();
    for (int s2 = 64; s2 > 0; s2 >>= 1) {
        if (tid < s2) red[tid] = fmaxf(red[tid], red[tid + s2]);
        __syncthreads();
    }
    m = red[0];
    __syncthreads();

    float sum = 0.0f;
    const bool allmasked = !(m > -__builtin_inff());
    for (int kk = tid; kk < L; kk += 128) {
        const float e = allmasked ? 0.0f : __expf(sc[kk] - m);
        sc[kk] = e;
        sum += e;
    }
    red[tid] = sum; __syncthreads();
    for (int s2 = 64; s2 > 0; s2 >>= 1) {
        if (tid < s2) red[tid] += red[tid + s2];
        __syncthreads();
    }
    sum = red[0];
    const float inv = (sum > 0.0f) ? 1.0f / sum : 0.0f;
    __syncthreads();

    const int d = tid;
    const float* vbase = v + (long)(b * L) * (H_ * DK_) + h * DK_ + d;
    float accd = 0.0f;
    for (int kk = 0; kk < L; ++kk) accd += sc[kk] * vbase[(long)kk * (H_ * DK_)];
    o[rowQ + d] = accd * inv;
}

// ---------------------------------------------------------------------------
// LayerNorm over D=256, one wave per row (8 elems/lane), wave32 shuffles.
// ---------------------------------------------------------------------------
__global__ __launch_bounds__(256)
void layernorm_kernel(const float* __restrict__ x, const float* __restrict__ g,
                      const float* __restrict__ bta, const float* __restrict__ rowmask,
                      float* __restrict__ y, int M)
{
    const int wave = threadIdx.x >> 5;
    const int lane = threadIdx.x & 31;
    const int row  = blockIdx.x * 8 + wave;
    if (row >= M) return;

    const float* xr = x + (long)row * D_;
    float vals[8];
    float s = 0.0f;
    #pragma unroll
    for (int i = 0; i < 8; ++i) { vals[i] = xr[lane * 8 + i]; s += vals[i]; }
    #pragma unroll
    for (int off = 16; off > 0; off >>= 1) s += __shfl_xor(s, off, 32);
    const float mean = s * (1.0f / D_);

    float vs = 0.0f;
    #pragma unroll
    for (int i = 0; i < 8; ++i) { const float dd = vals[i] - mean; vs += dd * dd; }
    #pragma unroll
    for (int off = 16; off > 0; off >>= 1) vs += __shfl_xor(vs, off, 32);
    const float r = rsqrtf(vs * (1.0f / D_) + 1e-5f);

    const float mask = rowmask ? rowmask[row] : 1.0f;
    float* yr = y + (long)row * D_;
    #pragma unroll
    for (int i = 0; i < 8; ++i) {
        const int c = lane * 8 + i;
        yr[c] = ((vals[i] - mean) * r * g[c] + bta[c]) * mask;
    }
}

// ---------------------------------------------------------------------------
// Encoder embedding + pad mask: x = word_emb[seq] + pos_emb[pos]
// ---------------------------------------------------------------------------
__global__ __launch_bounds__(256)
void enc_embed_kernel(const int* __restrict__ src_seq, const int* __restrict__ src_pos,
                      const float* __restrict__ word_emb, const float* __restrict__ pos_emb,
                      float* __restrict__ x, float* __restrict__ rowmask)
{
    const int row = blockIdx.x;
    const int c   = threadIdx.x;
    const int w = src_seq[row];
    const int p = src_pos[row];
    x[(long)row * D_ + c] = word_emb[(long)w * D_ + c] + pos_emb[(long)p * D_ + c];
    if (c == 0) rowmask[row] = (w != 0) ? 1.0f : 0.0f;
}

// ---------------------------------------------------------------------------
// Length regulator: per-batch cumsum of durations -> mel->src gather map.
// ---------------------------------------------------------------------------
__global__ __launch_bounds__(256)
void regulator_kernel(const int* __restrict__ durations, int* __restrict__ map,
                      int* __restrict__ mel_len)
{
    __shared__ int ends[LSRC_ + 1];
    const int b = blockIdx.x;
    const int tid = threadIdx.x;
    for (int m = tid; m < MMAX_; m += 256) map[b * MMAX_ + m] = -1;
    if (tid == 0) {
        int acc = 0;
        ends[0] = 0;
        for (int l = 0; l < LSRC_; ++l) { acc += durations[b * LSRC_ + l]; ends[l + 1] = acc; }
        mel_len[b] = acc > MMAX_ ? MMAX_ : acc;
    }
    __syncthreads();
    for (int l = tid; l < LSRC_; l += 256) {
        const int s = ends[l];
        int e = ends[l + 1];
        if (e > MMAX_) e = MMAX_;
        for (int m = s; m < e; ++m) map[b * MMAX_ + m] = b * LSRC_ + l;
    }
}

// ---------------------------------------------------------------------------
// Decoder embedding: y = gather(x_enc, map) + pos_emb[mel_pos], + pad mask.
// ---------------------------------------------------------------------------
__global__ __launch_bounds__(256)
void dec_embed_kernel(const float* __restrict__ xenc, const float* __restrict__ pos_emb,
                      const int* __restrict__ map, const int* __restrict__ mel_len,
                      float* __restrict__ y, float* __restrict__ rowmask)
{
    const int row = blockIdx.x;               // 0 .. B*MMAX-1
    const int c   = threadIdx.x;
    const int b = row / MMAX_;
    const int m = row % MMAX_;
    const int src = map[row];
    const float val = (src >= 0) ? xenc[(long)src * D_ + c] : 0.0f;
    const int mp = (m < mel_len[b]) ? (m + 1) : 0;
    y[(long)row * D_ + c] = val + pos_emb[(long)mp * D_ + c];
    if (c == 0) rowmask[row] = (mp != 0) ? 1.0f : 0.0f;
}

// ---------------------------------------------------------------------------
// Host orchestration
// ---------------------------------------------------------------------------
struct StackP {
    const float *wq, *bq, *wk, *bk, *wv, *bv, *wo, *bo;
    const float *ln1g, *ln1b, *ln2g, *ln2b;
    const float *c1w, *c1b, *c2w, *c2b;
};

static void gemm(hipStream_t s, const float* A, int lda, int Lrows,
                 const float* Bm, long bs_k, long bs_n,
                 const float* bias, const float* resid, const float* rowmask,
                 float* C, int ldc, int M, int N, int K,
                 int ntaps, int tapShift0, long tapBStep, bool relu)
{
    dim3 grid(M / 64, (N + 63) / 64);
    if (relu) {
        if (bs_n == 1)
            gemm_wmma_kernel<true, true><<<grid, 128, 0, s>>>(A, lda, Lrows, Bm, bs_k, bs_n,
                bias, resid, rowmask, C, ldc, N, K, ntaps, tapShift0, tapBStep);
        else
            gemm_wmma_kernel<true, false><<<grid, 128, 0, s>>>(A, lda, Lrows, Bm, bs_k, bs_n,
                bias, resid, rowmask, C, ldc, N, K, ntaps, tapShift0, tapBStep);
    } else {
        if (bs_n == 1)
            gemm_wmma_kernel<false, true><<<grid, 128, 0, s>>>(A, lda, Lrows, Bm, bs_k, bs_n,
                bias, resid, rowmask, C, ldc, N, K, ntaps, tapShift0, tapBStep);
        else
            gemm_wmma_kernel<false, false><<<grid, 128, 0, s>>>(A, lda, Lrows, Bm, bs_k, bs_n,
                bias, resid, rowmask, C, ldc, N, K, ntaps, tapShift0, tapBStep);
    }
}

static void run_stack(hipStream_t s, const StackP& p, float* x, int Mrows, int Lrows,
                      const float* mask, float* xn, float* qb, float* kb, float* vb,
                      float* ob, float* yb, float* hb)
{
    for (int i = 0; i < NL_; ++i) {
        const float* wq = p.wq + (long)i * D_ * (H_ * DK_);
        const float* wk = p.wk + (long)i * D_ * (H_ * DK_);
        const float* wv = p.wv + (long)i * D_ * (H_ * DK_);
        const float* wo = p.wo + (long)i * (H_ * DK_) * D_;
        const float* bq = p.bq + (long)i * (H_ * DK_);
        const float* bk = p.bk + (long)i * (H_ * DK_);
        const float* bv = p.bv + (long)i * (H_ * DK_);
        const float* bo = p.bo + (long)i * D_;
        const float* l1g = p.ln1g + (long)i * D_, *l1b = p.ln1b + (long)i * D_;
        const float* l2g = p.ln2g + (long)i * D_, *l2b = p.ln2b + (long)i * D_;
        const float* c1w = p.c1w + (long)i * FF_ * D_ * K1_;
        const float* c1b = p.c1b + (long)i * FF_;
        const float* c2w = p.c2w + (long)i * D_ * FF_;
        const float* c2b = p.c2b + (long)i * D_;

        // xn = LN1(x)
        layernorm_kernel<<<(Mrows + 7) / 8, 256, 0, s>>>(x, l1g, l1b, nullptr, xn, Mrows);
        // q/k/v = xn @ w + b   (weights D x 256: bs_k = 256, bs_n = 1)
        gemm(s, xn, D_, 0, wq, H_ * DK_, 1, bq, nullptr, nullptr, qb, H_ * DK_,
             Mrows, H_ * DK_, D_, 1, 0, 0, false);
        gemm(s, xn, D_, 0, wk, H_ * DK_, 1, bk, nullptr, nullptr, kb, H_ * DK_,
             Mrows, H_ * DK_, D_, 1, 0, 0, false);
        gemm(s, xn, D_, 0, wv, H_ * DK_, 1, bv, nullptr, nullptr, vb, H_ * DK_,
             Mrows, H_ * DK_, D_, 1, 0, 0, false);
        // attention -> ob
        attention_kernel<<<B_ * H_ * Lrows, 128, 0, s>>>(qb, kb, vb, mask, ob, Lrows);
        // xn = ob @ wo + bo + xn   (output aliases resid; element-wise safe)
        gemm(s, ob, H_ * DK_, 0, wo, D_, 1, bo, xn, nullptr, xn, D_,
             Mrows, D_, H_ * DK_, 1, 0, 0, false);
        // yb = LN2(xn) * mask
        layernorm_kernel<<<(Mrows + 7) / 8, 256, 0, s>>>(xn, l2g, l2b, mask, yb, Mrows);
        // hb = relu(conv1(yb)): 9 taps, B element (k=c, n=f) at f*D*K1 + c*K1 + t
        gemm(s, yb, D_, Lrows, c1w, K1_, (long)D_ * K1_, c1b, nullptr, nullptr,
             hb, FF_, Mrows, FF_, D_, K1_, -P1_, 1, true);
        // x = (conv2(hb) + c2b + yb) * mask: B element (k=f, n=d) at d*FF + f
        gemm(s, hb, FF_, 0, c2w, 1, FF_, c2b, yb, mask, x, D_,
             Mrows, D_, FF_, 1, 0, 0, false);
    }
}

extern "C" void kernel_launch(void* const* d_in, const int* in_sizes, int n_in,
                              void* d_out, int out_size, void* d_ws, size_t ws_size,
                              hipStream_t stream)
{
    (void)in_sizes; (void)n_in; (void)out_size; (void)ws_size;

    // setup_inputs() order; param dicts flattened in _stack_params insertion
    // order: wq,bq,wk,bk,wv,bv,wo,bo,ln1_g,ln1_b,ln2_g,ln2_b,c1_w,c1_b,c2_w,c2_b
    const float* word_emb = (const float*)d_in[0];
    const float* pos_emb  = (const float*)d_in[1];
    auto getp = [&](int base) {
        StackP P;
        P.wq   = (const float*)d_in[base + 0];  P.bq   = (const float*)d_in[base + 1];
        P.wk   = (const float*)d_in[base + 2];  P.bk   = (const float*)d_in[base + 3];
        P.wv   = (const float*)d_in[base + 4];  P.bv   = (const float*)d_in[base + 5];
        P.wo   = (const float*)d_in[base + 6];  P.bo   = (const float*)d_in[base + 7];
        P.ln1g = (const float*)d_in[base + 8];  P.ln1b = (const float*)d_in[base + 9];
        P.ln2g = (const float*)d_in[base + 10]; P.ln2b = (const float*)d_in[base + 11];
        P.c1w  = (const float*)d_in[base + 12]; P.c1b  = (const float*)d_in[base + 13];
        P.c2w  = (const float*)d_in[base + 14]; P.c2b  = (const float*)d_in[base + 15];
        return P;
    };
    StackP enc = getp(2);
    StackP dec = getp(18);
    const float* mel_w = (const float*)d_in[34];
    const float* mel_b = (const float*)d_in[35];
    const int* src_seq   = (const int*)d_in[36];
    const int* src_pos   = (const int*)d_in[37];
    const int* durations = (const int*)d_in[38];
    float* out = (float*)d_out;

    // ---- workspace carve-up (floats) -------------------------------------
    float* ws = (float*)d_ws;
    float* x_enc  = ws;                               // ROWS_ENC * 256
    float* x_dec  = x_enc + (long)ROWS_ENC * D_;      // ROWS_DEC * 256
    float* xn     = x_dec + (long)ROWS_DEC * D_;      // ROWS_DEC * 256
    float* yb     = xn    + (long)ROWS_DEC * D_;      // ROWS_DEC * 256
    float* qb     = yb    + (long)ROWS_DEC * D_;      // 4 * ROWS_DEC * 256 (q,k,v,o)
    float* kb     = qb + (long)ROWS_DEC * D_;
    float* vb     = kb + (long)ROWS_DEC * D_;
    float* ob     = vb + (long)ROWS_DEC * D_;
    float* hb     = qb;                               // alias: ROWS_DEC * FF == 4*ROWS_DEC*D
    float* mask_e = ob + (long)ROWS_DEC * D_;         // ROWS_ENC
    float* mask_d = mask_e + ROWS_ENC;                // ROWS_DEC
    int*   map    = (int*)(mask_d + ROWS_DEC);        // B * MMAX
    int*   melLen = map + B_ * MMAX_;                 // B

    // ---- encoder ---------------------------------------------------------
    enc_embed_kernel<<<ROWS_ENC, 256, 0, stream>>>(src_seq, src_pos, word_emb,
                                                   pos_emb, x_enc, mask_e);
    run_stack(stream, enc, x_enc, ROWS_ENC, LSRC_, mask_e,
              xn, qb, kb, vb, ob, yb, hb);

    // ---- length regulator + decoder embedding ----------------------------
    regulator_kernel<<<B_, 256, 0, stream>>>(durations, map, melLen);
    dec_embed_kernel<<<ROWS_DEC, 256, 0, stream>>>(x_enc, pos_emb, map, melLen,
                                                   x_dec, mask_d);

    // ---- decoder ---------------------------------------------------------
    run_stack(stream, dec, x_dec, ROWS_DEC, MMAX_, mask_d,
              xn, qb, kb, vb, ob, yb, hb);

    // ---- mel head: out = x_dec @ mel_w + mel_b ---------------------------
    gemm(stream, x_dec, D_, 0, mel_w, NMELS_, 1, mel_b, nullptr, nullptr,
         out, NMELS_, ROWS_DEC, NMELS_, D_, 1, 0, 0, false);
}
